// SelfAttention_39402029974037
// MI455X (gfx1250) — matrix operand order repaired and checked by reading, not compiled, not run on hardware
//
#include <hip/hip_runtime.h>

// ---------------------------------------------------------------------------
// Self-attention on gfx1250 (MI455X): bf16 WMMA everywhere.
//   Kernel 1: q/v = X @ W^T + b (row-major bf16), k stored d-major (K^T),
//             register-pipelined global loads.
//   Kernel 2: flash attention, online softmax, fp32 accumulate,
//             double-buffered GLOBAL_LOAD_ASYNC_TO_LDS staging (ASYNCcnt).
// ---------------------------------------------------------------------------

typedef __attribute__((ext_vector_type(2)))  float  v2f;
typedef __attribute__((ext_vector_type(2)))  __bf16 v2bf;
typedef __attribute__((ext_vector_type(8)))  __bf16 v8bf;
typedef __attribute__((ext_vector_type(16))) __bf16 v16bf;
typedef __attribute__((ext_vector_type(8)))  float  v8f;
typedef __attribute__((ext_vector_type(4)))  int    v4i;

#define NB    4
#define TSEQ  4096
#define TT    16384          // NB * TSEQ
#define DIM   1024
#define LOG2E 1.4426950408889634f

union U8bf { v8bf v; v2bf p[4]; };
union F4   { float4 v; float f[4]; };

// ---------------- bf16 conversion helpers (native fptrunc, RTNE) ---------
__device__ __forceinline__ __bf16 f2bf(float f) { return (__bf16)f; }

__device__ __forceinline__ v2bf pk2(float a, float b) {
#if __has_builtin(__builtin_amdgcn_cvt_pk_bf16_f32)
  return __builtin_amdgcn_cvt_pk_bf16_f32(a, b);   // v_cvt_pk_bf16_f32
#else
  v2f t; t[0] = a; t[1] = b;
  return __builtin_convertvector(t, v2bf);         // vector fptrunc
#endif
}

// ---------------- async global -> LDS copy (gfx1250) ----------------
typedef __attribute__((address_space(1))) v4i* gv4ip;
typedef __attribute__((address_space(3))) v4i* lv4ip;

#if __has_builtin(__builtin_amdgcn_global_load_async_to_lds_b128)
#define HAVE_ASYNC_LDS 1
__device__ __forceinline__ void async_cp16(void* lds, const void* g) {
  __builtin_amdgcn_global_load_async_to_lds_b128(
      (gv4ip)(v4i*)g, (lv4ip)(v4i*)lds, 0, 0);
}
#else
#define HAVE_ASYNC_LDS 0
__device__ __forceinline__ void async_cp16(void* lds, const void* g) {
  *(uint4*)lds = *(const uint4*)g;                 // fallback: through VGPRs
}
#endif

__device__ __forceinline__ void wait_async_lds() {
#if HAVE_ASYNC_LDS
#if __has_builtin(__builtin_amdgcn_s_wait_asynccnt)
  __builtin_amdgcn_s_wait_asynccnt(0);
#else
  asm volatile("s_wait_asynccnt 0" ::: "memory");
#endif
#endif
}

// ---------------- WMMA fragment loads (documented VGPR layouts) ----------
// A fragment (16x32, lane = M row): lane<16 holds K {0..7,16..23},
// lane>=16 holds K {8..15,24..31}. rowbase = &[row][k0]; h = lane>>4.
__device__ __forceinline__ v16bf bfcat(v8bf lo, v8bf hi) {
  return __builtin_shufflevector(lo, hi, 0,1,2,3,4,5,6,7,8,9,10,11,12,13,14,15);
}
__device__ __forceinline__ v16bf fragA(const __bf16* rowbase, int h) {
  v8bf lo = *(const v8bf*)(rowbase + h * 8);
  v8bf hi = *(const v8bf*)(rowbase + 16 + h * 8);
  return bfcat(lo, hi);
}
// B fragment (32x16, lane = N column, K {0..7,16..23} / {8..15,24..31} by h, row stride ld).
__device__ __forceinline__ v16bf fragB(const __bf16* p, int ld, int h) {
  v16bf r;
#pragma unroll
  for (int s = 0; s < 8; s++) {
    r[s]     = p[(h * 8 + s) * ld];
    r[s + 8] = p[(16 + h * 8 + s) * ld];
  }
  return r;
}

// ---------------------------------------------------------------------------
// Kernel 1: fused QKV projection GEMM, register-pipelined staging.
// Grid: (TT/128, 24). blockIdx.y: [0..7]=q cols, [8..15]=k, [16..23]=v.
// Workgroup: 256 threads = 8 waves. 128x128 tile; wave = 32 rows x 64 cols.
// q, v stored row-major [t][d]; k stored transposed per batch: kT[n][d][t].
// ---------------------------------------------------------------------------
__global__ __launch_bounds__(256) void qkv_kernel(
    const float* __restrict__ X,
    const float* __restrict__ Wq, const float* __restrict__ Wk,
    const float* __restrict__ Wv,
    const float* __restrict__ bq, const float* __restrict__ bk,
    const float* __restrict__ bv,
    __bf16* __restrict__ ws)
{
  __shared__ __align__(16) __bf16 Xs[128 * 32];   // row-major [row][k]
  __shared__ __align__(16) __bf16 WsT[32 * 128];  // k-major  [k][n]

  const int tid  = threadIdx.x;
  const int lane = tid & 31;
  const int wid  = tid >> 5;
  const int m    = lane & 15;
  const int h    = lane >> 4;

  const int r0    = blockIdx.x * 128;
  const int cb    = blockIdx.y;
  const int which = cb >> 3;              // 0=q 1=k 2=v
  const int o0    = (cb & 7) * 128;

  const float* W    = (which == 0) ? Wq : ((which == 1) ? Wk : Wv);
  const float* bias = (which == 0) ? bq : ((which == 1) ? bk : bv);

  const int wrow = (wid & 3) * 32;
  const int wcol = (wid >> 2) * 64;

  v8f acc[2][4];
#pragma unroll
  for (int i = 0; i < 2; i++)
#pragma unroll
    for (int j = 0; j < 4; j++) acc[i][j] = (v8f)(0.0f);

  // staging maps
  const int srow  = tid & 127;            // X row
  const int shalf = tid >> 7;             // 16-wide K half for X
  const int npair = tid & 63;             // W column pair (2*npair, 2*npair+1)
  const int kq    = tid >> 6;             // 0..3 -> 8 k's each for W

  const float* srcX  = X + (size_t)(r0 + srow) * DIM + shalf * 16;
  const float* srcW0 = W + (size_t)(o0 + 2 * npair) * DIM + kq * 8;
  const float* srcW1 = srcW0 + DIM;

  F4 rx[4], rwa[2], rwb[2];
  // prologue: load k-step 0 into registers
#pragma unroll
  for (int q = 0; q < 4; q++) rx[q].v = ((const float4*)srcX)[q];
#pragma unroll
  for (int q = 0; q < 2; q++) {
    rwa[q].v = ((const float4*)srcW0)[q];
    rwb[q].v = ((const float4*)srcW1)[q];
  }

  for (int k0 = 0; k0 < DIM; k0 += 32) {
    __syncthreads();
    // ---- store staged registers to LDS (packed bf16 converts) ----
    {
      U8bf t0, t1;
      t0.p[0] = pk2(rx[0].f[0], rx[0].f[1]); t0.p[1] = pk2(rx[0].f[2], rx[0].f[3]);
      t0.p[2] = pk2(rx[1].f[0], rx[1].f[1]); t0.p[3] = pk2(rx[1].f[2], rx[1].f[3]);
      t1.p[0] = pk2(rx[2].f[0], rx[2].f[1]); t1.p[1] = pk2(rx[2].f[2], rx[2].f[3]);
      t1.p[2] = pk2(rx[3].f[0], rx[3].f[1]); t1.p[3] = pk2(rx[3].f[2], rx[3].f[3]);
      *(v8bf*)&Xs[srow * 32 + shalf * 16 + 0] = t0.v;
      *(v8bf*)&Xs[srow * 32 + shalf * 16 + 8] = t1.v;
#pragma unroll
      for (int i = 0; i < 8; i++) {
        v2bf p = pk2(rwa[i / 4].f[i & 3], rwb[i / 4].f[i & 3]);
        *(v2bf*)&WsT[(kq * 8 + i) * 128 + 2 * npair] = p;
      }
    }
    __syncthreads();
    // ---- prefetch next k-step into registers (overlaps with WMMAs) ----
    if (k0 + 32 < DIM) {
#pragma unroll
      for (int q = 0; q < 4; q++)
        rx[q].v = ((const float4*)(srcX + k0 + 32))[q];
#pragma unroll
      for (int q = 0; q < 2; q++) {
        rwa[q].v = ((const float4*)(srcW0 + k0 + 32))[q];
        rwb[q].v = ((const float4*)(srcW1 + k0 + 32))[q];
      }
    }
    // ---- 8 WMMAs per wave per K step ----
    v16bf a[2], b[4];
#pragma unroll
    for (int i = 0; i < 2; i++)
      a[i] = fragA(&Xs[(wrow + i * 16 + m) * 32], h);
#pragma unroll
    for (int j = 0; j < 4; j++)
      b[j] = fragB(&WsT[wcol + j * 16 + m], 128, h);
#pragma unroll
    for (int i = 0; i < 2; i++)
#pragma unroll
      for (int j = 0; j < 4; j++)
        acc[i][j] = __builtin_amdgcn_wmma_f32_16x16x32_bf16(
            false, a[i], false, b[j], (short)0, acc[i][j], false, false);
  }

  // ---- epilogue: bias + bf16 store. C layout: row = r + 8h, col = m ----
  if (which == 1) {
    // K stored transposed: kT[n][d][t]; each lane writes 8 consecutive t's.
    const int nbat = r0 / TSEQ;
    const int t0   = r0 % TSEQ;
    __bf16* kT = ws + (size_t)TT * DIM + (size_t)nbat * DIM * TSEQ;
#pragma unroll
    for (int i = 0; i < 2; i++)
#pragma unroll
      for (int j = 0; j < 4; j++) {
        const int colg = o0 + wcol + j * 16 + m;
        const float bb = bias[colg];
        U8bf tmp;
#pragma unroll
        for (int r = 0; r < 8; r += 2)
          tmp.p[r / 2] = pk2(acc[i][j][r] + bb, acc[i][j][r + 1] + bb);
        const int rowl = t0 + wrow + i * 16 + 8 * h;
        *(v8bf*)(kT + (size_t)colg * TSEQ + rowl) = tmp.v;
      }
  } else {
    __bf16* dst = ws + (size_t)which * TT * DIM;   // which: 0=q, 2=v
#pragma unroll
    for (int i = 0; i < 2; i++)
#pragma unroll
      for (int j = 0; j < 4; j++) {
        const int colg = o0 + wcol + j * 16 + m;
        const float bb = bias[colg];
#pragma unroll
        for (int r = 0; r < 8; r++) {
          const int rowg = r0 + wrow + i * 16 + r + 8 * h;
          dst[(size_t)rowg * DIM + colg] = f2bf(acc[i][j][r] + bb);
        }
      }
  }
}

// ---------------------------------------------------------------------------
// Kernel 2: flash attention. Grid: TT/16 = 1024 workgroups of 256 threads.
// 16 queries per workgroup; 32-key steps; wave w owns V columns [128w,128w+128).
// K^T and V tiles double-buffered via GLOBAL_LOAD_ASYNC_TO_LDS: tile j+1 is
// issued right after the consuming barrier and completes while tile j's
// S/softmax/PV run. LDS: ~291 KB of the 320 KB WGP pool.
// ---------------------------------------------------------------------------
__global__ __launch_bounds__(256) void attn_kernel(
    const __bf16* __restrict__ qws, const __bf16* __restrict__ kTws,
    const __bf16* __restrict__ vws, float* __restrict__ out)
{
  __shared__ __align__(16) __bf16 qs[16 * DIM];        // 32 KB  [qrow][d]
  __shared__ __align__(16) __bf16 ksT[2][DIM * 32];    // 128 KB [buf][d][key]
  __shared__ __align__(16) __bf16 vsm[2][32 * DIM];    // 128 KB [buf][key][d]
  __shared__ __align__(16) float  sred[16 * 32];
  __shared__ __align__(16) __bf16 pls[16 * 32];
  __shared__ float corr_l[16];
  __shared__ float invl[16];

  const int tid  = threadIdx.x;
  const int lane = tid & 31;
  const int wid  = tid >> 5;
  const int m    = lane & 15;
  const int h    = lane >> 4;

  const int rowg0 = blockIdx.x * 16;
  const int nbat  = rowg0 / TSEQ;
  const int krow0 = nbat * TSEQ;                         // v rows base
  const __bf16* kT = kTws + (size_t)nbat * DIM * TSEQ;   // [d][t]

  // issue all async loads for key-tile jj into buffer bb
  auto stage_tile = [&](int jj, int bb) {
    const int key0 = jj * 32;
#pragma unroll
    for (int i = 0; i < 16; i++) {                       // K^T: 1024 rows x 64B
      const int idx = i * 256 + tid;
      const int d = idx >> 2, c = (idx & 3) * 8;
      async_cp16(&ksT[bb][d * 32 + c], kT + (size_t)d * TSEQ + key0 + c);
    }
#pragma unroll
    for (int i = 0; i < 16; i++) {                       // V: straight 64 KB
      const int idx = i * 256 + tid;
      async_cp16(&vsm[bb][idx * 8],
                 vws + (size_t)(krow0 + key0) * DIM + idx * 8);
    }
  };

  // ---- prologue: q tile + first key tile in flight ----
#pragma unroll
  for (int i = 0; i < 8; i++) {
    const int idx = i * 256 + tid;                       // 2048 x 16B
    async_cp16(&qs[idx * 8], qws + (size_t)rowg0 * DIM + idx * 8);
  }
  stage_tile(0, 0);
  for (int i = tid; i < 16 * 32; i += 256) sred[i] = 0.0f;

  float mrow = -1e30f, lrow = 0.0f;                      // valid for tid < 16

  v8f oacc[8];
#pragma unroll
  for (int t = 0; t < 8; t++) oacc[t] = (v8f)(0.0f);

  const int vcol0 = wid * 128;

  for (int j = 0; j < TSEQ / 32; j++) {
    const int bb = j & 1;
    wait_async_lds();          // my loads for tile j (and q on j==0) done
    __syncthreads();           // everyone's tile j ready; buffers bb^1 free
    if (j + 1 < TSEQ / 32) stage_tile(j + 1, bb ^ 1);   // overlap with compute

    // ---- scores: wave wid covers d slice [128*wid, 128*wid+128) ----
    {
      v8f c2[2];
      c2[0] = (v8f)(0.0f); c2[1] = (v8f)(0.0f);
#pragma unroll
      for (int kk = 0; kk < 4; kk++) {
        const int kb = wid * 128 + kk * 32;
        v16bf a = fragA(&qs[m * DIM + kb], h);
#pragma unroll
        for (int jj = 0; jj < 2; jj++) {
          v16bf b = fragB(&ksT[bb][kb * 32 + jj * 16 + m], 32, h);
          c2[jj] = __builtin_amdgcn_wmma_f32_16x16x32_bf16(
              false, a, false, b, (short)0, c2[jj], false, false);
        }
      }
#pragma unroll
      for (int jj = 0; jj < 2; jj++)
#pragma unroll
        for (int r = 0; r < 8; r++)
          atomicAdd(&sred[(r + 8 * h) * 32 + jj * 16 + m], c2[jj][r]);
    }
    __syncthreads();

    // ---- online softmax: one thread per query row ----
    if (tid < 16) {
      float s[32];
      float mx = mrow;
#pragma unroll
      for (int c = 0; c < 32; c++) {
        s[c] = sred[tid * 32 + c] * 0.03125f;            // 1/sqrt(1024)
        mx = fmaxf(mx, s[c]);
      }
      const float cr = exp2f((mrow - mx) * LOG2E);
      float sum = 0.0f;
#pragma unroll
      for (int c = 0; c < 32; c += 2) {
        const float p0 = exp2f((s[c] - mx) * LOG2E);
        const float p1 = exp2f((s[c + 1] - mx) * LOG2E);
        sum += p0 + p1;
        *(v2bf*)&pls[tid * 32 + c] = pk2(p0, p1);
        sred[tid * 32 + c]     = 0.0f;                   // re-zero for next step
        sred[tid * 32 + c + 1] = 0.0f;
      }
      lrow = lrow * cr + sum;
      mrow = mx;
      corr_l[tid] = cr;
    }
    __syncthreads();

    // ---- P @ V: rescale accumulators, then 8 WMMAs per wave ----
    {
      float cf[8];
#pragma unroll
      for (int r = 0; r < 8; r++) cf[r] = corr_l[r + 8 * h];
      v16bf a = fragA(&pls[m * 32], h);
#pragma unroll
      for (int t = 0; t < 8; t++) {
#pragma unroll
        for (int r = 0; r < 8; r++) oacc[t][r] *= cf[r];
        v16bf b = fragB(&vsm[bb][vcol0 + t * 16 + m], DIM, h);
        oacc[t] = __builtin_amdgcn_wmma_f32_16x16x32_bf16(
            false, a, false, b, (short)0, oacc[t], false, false);
      }
    }
  }

  __syncthreads();
  if (tid < 16) invl[tid] = 1.0f / lrow;
  __syncthreads();

  // ---- final normalize + fp32 store ----
#pragma unroll
  for (int t = 0; t < 8; t++) {
    const int colg = vcol0 + t * 16 + m;
#pragma unroll
    for (int r = 0; r < 8; r++) {
      const int rowg = rowg0 + r + 8 * h;
      out[(size_t)rowg * DIM + colg] = oacc[t][r] * invl[r + 8 * h];
    }
  }
}

// ---------------------------------------------------------------------------
extern "C" void kernel_launch(void* const* d_in, const int* in_sizes, int n_in,
                              void* d_out, int out_size, void* d_ws,
                              size_t ws_size, hipStream_t stream)
{
  (void)in_sizes; (void)n_in; (void)out_size; (void)ws_size;
  const float* X  = (const float*)d_in[0];
  const float* Wq = (const float*)d_in[1];
  const float* bq = (const float*)d_in[2];
  const float* Wk = (const float*)d_in[3];
  const float* bk = (const float*)d_in[4];
  const float* Wv = (const float*)d_in[5];
  const float* bv = (const float*)d_in[6];

  __bf16* ws   = (__bf16*)d_ws;            // q | kT | v, bf16, TT*DIM each
  __bf16* ws_q = ws;
  __bf16* ws_k = ws + (size_t)TT * DIM;    // kT: per-batch [DIM][TSEQ]
  __bf16* ws_v = ws + (size_t)2 * TT * DIM;

  qkv_kernel<<<dim3(TT / 128, 24), 256, 0, stream>>>(
      X, Wq, Wk, Wv, bq, bk, bv, ws);
  attn_kernel<<<dim3(TT / 16), 256, 0, stream>>>(
      ws_q, ws_k, ws_v, (float*)d_out);
}
